// DynamicCoeffHeteroGraphSAGE_61151744361084
// MI455X (gfx1250) — matrix operand order repaired and checked by reading, not compiled, not run on hardware
//
#include <hip/hip_runtime.h>
#include <hip/hip_bf16.h>

// ---------------------------------------------------------------------------
// DynamicCoeffHeteroGraphSAGE for MI455X (gfx1250, wave32, WMMA)
//
// Pipeline per call:
//   1. schema_kernel (1 block): h = schema_x@pre_W.T+b (-> ori), tiny GCN -> sf
//   2. per layer l in {0,1}:
//      a. weight_kernel: coeffs = [sf_s,sf_d]@coeff_W.T+b ; W_et = sum coeffs*bases
//      b. zero msg/cnt buffers
//      c. 3x scatter_kernel: segment-sum gathers (float4 + 4x atomicAdd)
//      d. sage_gemm_ln_kernel<1> (dst=b) and <2> (dst=a):
//         bf16 WMMA GEMM of (msg/max(cnt,1)+x_dst) @ W.T, fused bias+LN+ReLU
// ---------------------------------------------------------------------------

#define NA_N   50000
#define NB_N   50000
#define C_N    128
#define E_N    800000
#define S_N    8
#define IN_N   64
#define H_N    64
#define NBASES 8
#define EPS_F  1e-5f

typedef __attribute__((ext_vector_type(16))) __bf16 v16bf;
typedef __attribute__((ext_vector_type(8)))  float  v8f;

// ------------------------------ zero ---------------------------------------
__global__ __launch_bounds__(256) void zero_kernel(float* __restrict__ p, long n) {
  long i = (long)blockIdx.x * blockDim.x + threadIdx.x;
  long stride = (long)gridDim.x * blockDim.x;
  for (; i < n; i += stride) p[i] = 0.0f;
}

// --------------------------- schema GCN (tiny) ------------------------------
__global__ __launch_bounds__(256) void schema_kernel(
    const float* __restrict__ sx, const int* __restrict__ sei,
    const float* __restrict__ preW, const float* __restrict__ preb,
    const float* __restrict__ gcnW, const float* __restrict__ gcnb,
    float* __restrict__ sf_ws, float* __restrict__ out_sf, float* __restrict__ out_ori)
{
  __shared__ float h[S_N * H_N], xw[S_N * H_N], agg[S_N * H_N];
  __shared__ float deg[S_N], dinv[S_N];
  int t = threadIdx.x;
  // h = schema_x @ pre_W.T + pre_b ; ori = h
  for (int idx = t; idx < S_N * H_N; idx += 256) {
    int s = idx >> 6, j = idx & 63;
    float a = preb[j];
    for (int k = 0; k < IN_N; ++k) a += sx[s * IN_N + k] * preW[j * IN_N + k];
    h[idx] = a;
    out_ori[idx] = a;
  }
  if (t < S_N) deg[t] = 1.0f;  // self loop
  __syncthreads();
  if (t < 24) atomicAdd(&deg[sei[24 + t]], 1.0f);
  __syncthreads();
  if (t < S_N) dinv[t] = rsqrtf(fmaxf(deg[t], 1e-12f));
  // xw = h @ gcn_W.T ; agg = 0
  for (int idx = t; idx < S_N * H_N; idx += 256) {
    int s = idx >> 6, j = idx & 63;
    float a = 0.0f;
    for (int k = 0; k < H_N; ++k) a += h[s * H_N + k] * gcnW[j * H_N + k];
    xw[idx] = a;
    agg[idx] = 0.0f;
  }
  __syncthreads();
  // normalized scatter over 24 edges
  for (int idx = t; idx < 24 * H_N; idx += 256) {
    int e = idx >> 6, j = idx & 63;
    int s = sei[e], d = sei[24 + e];
    atomicAdd(&agg[d * H_N + j], xw[s * H_N + j] * dinv[s] * dinv[d]);
  }
  // self loops
  for (int idx = t; idx < S_N * H_N; idx += 256) {
    int s = idx >> 6;
    atomicAdd(&agg[idx], xw[idx] * dinv[s] * dinv[s]);
  }
  __syncthreads();
  for (int idx = t; idx < S_N * H_N; idx += 256) {
    int j = idx & 63;
    float v = fmaxf(agg[idx] + gcnb[j], 0.0f);
    sf_ws[idx] = v;
    out_sf[idx] = v;
  }
}

// ------------------- per-edge-type weight matrices --------------------------
// blockIdx.x = edge type: 0=ab (sf0,sf1), 1=ba (sf1,sf0), 2=aa (sf0,sf0)
__global__ __launch_bounds__(256) void weight_kernel(
    const float* __restrict__ sf, const float* __restrict__ coeffW,
    const float* __restrict__ coeffb, const float* __restrict__ bases_l,
    float* __restrict__ Wout)
{
  __shared__ float c[NBASES];
  int et = blockIdx.x;
  int srow = (et == 1) ? 1 : 0;
  int drow = (et == 0) ? 1 : 0;
  int t = threadIdx.x;
  if (t < NBASES) {
    float a = coeffb[t];
    for (int k = 0; k < H_N; ++k) a += sf[srow * H_N + k] * coeffW[t * 128 + k];
    for (int k = 0; k < H_N; ++k) a += sf[drow * H_N + k] * coeffW[t * 128 + 64 + k];
    c[t] = a;
  }
  __syncthreads();
  for (int idx = t; idx < C_N * C_N; idx += 256) {
    float a = 0.0f;
    #pragma unroll
    for (int i = 0; i < NBASES; ++i) a += c[i] * bases_l[i * (C_N * C_N) + idx];
    Wout[et * (C_N * C_N) + idx] = a;
  }
}

// ---------------------- edge scatter (segment sum) --------------------------
__global__ __launch_bounds__(256) void scatter_kernel(
    const float* __restrict__ xsrc, const int* __restrict__ ei,
    float* __restrict__ msg, float* __restrict__ cnt)
{
  long idx = (long)blockIdx.x * blockDim.x + threadIdx.x;
  int e = (int)(idx >> 5);
  int g = (int)(idx & 31);
  if (e >= E_N) return;
  int s = ei[e];
  int d = ei[E_N + e];
  const float4 v = *(const float4*)(xsrc + (size_t)s * C_N + g * 4);
  float* p = msg + (size_t)d * C_N + g * 4;
  atomicAdd(p + 0, v.x);
  atomicAdd(p + 1, v.y);
  atomicAdd(p + 2, v.z);
  atomicAdd(p + 3, v.w);
  if (g == 0) atomicAdd(cnt + d, 1.0f);
}

// ------------- fused SAGE GEMM + bias + LayerNorm + ReLU (WMMA) -------------
// Each wave computes a 16-row x 128-col tile of out = relu(LN(sum_cv Z_cv@W_cv.T + b)).
// Z = msg/max(cnt,1) + x_dst.  A/B in bf16, accumulate f32 (v_wmma_f32_16x16x32_bf16).
template <int NCONV>
__global__ __launch_bounds__(256) void sage_gemm_ln_kernel(
    const float* __restrict__ msg0, const float* __restrict__ cnt0,
    const float* __restrict__ W0,   const float* __restrict__ b0,
    const float* __restrict__ msg1, const float* __restrict__ cnt1,
    const float* __restrict__ W1,   const float* __restrict__ b1,
    const float* __restrict__ xdst, const float* __restrict__ lnw,
    const float* __restrict__ lnb,  float* __restrict__ out, int nrows)
{
  // Pre-swizzled B fragments: [conv][kchunk][coltile][lane][16 bf16]
  __shared__ __align__(32) __bf16 Bf[NCONV][4][8][32][16];
  const float* Ws[2] = { W0, W1 };
  int t = threadIdx.x;
  const int tot = NCONV * 4 * 8 * 32 * 16;
  for (int idx = t; idx < tot; idx += 256) {
    int i16  = idx & 15;
    int lane = (idx >> 4) & 31;
    int ct   = (idx >> 9) & 7;
    int kc4  = (idx >> 12) & 3;
    int cv   = idx >> 14;
    int half = lane >> 4, lr = lane & 15;
    int col  = ct * 16 + lr;
    // B interleaved layout: half 0 -> K = kc+0..7 and kc+16..23; half 1 -> +8
    int k = kc4 * 32 + half * 8 + (i16 & 7) + (i16 >> 3) * 16;
    Bf[cv][kc4][ct][lane][i16] = (__bf16)Ws[cv][col * C_N + k];  // B[k][n] = W[n][k]
  }
  __syncthreads();

  int wave = t >> 5, lane = t & 31;
  int tile = blockIdx.x * 8 + wave;
  if (tile >= (nrows >> 4)) return;
  int half = lane >> 4, lr = lane & 15;
  int m0 = tile << 4;
  int row = m0 + lr;  // A-matrix row for this lane (both halves cover M=0..15)

  v8f acc[8];
  #pragma unroll
  for (int ct = 0; ct < 8; ++ct) acc[ct] = (v8f){0, 0, 0, 0, 0, 0, 0, 0};

  const float* msgs[2] = { msg0, msg1 };
  const float* cnts[2] = { cnt0, cnt1 };
  #pragma unroll
  for (int cv = 0; cv < NCONV; ++cv) {
    float rcp = 1.0f / fmaxf(cnts[cv][row], 1.0f);
    const float* mrow = msgs[cv] + (size_t)row * C_N;
    const float* xrow = xdst + (size_t)row * C_N;
    #pragma unroll
    for (int kc4 = 0; kc4 < 4; ++kc4) {
      // A interleaved layout: half 0 -> K = kc+0..7 and kc+16..23; half 1 -> +8
      int kb0 = kc4 * 32 + half * 8;
      int kb1 = kb0 + 16;
      float4 ma0 = *(const float4*)(mrow + kb0);
      float4 ma1 = *(const float4*)(mrow + kb0 + 4);
      float4 mb0 = *(const float4*)(mrow + kb1);
      float4 mb1 = *(const float4*)(mrow + kb1 + 4);
      float4 xa0 = *(const float4*)(xrow + kb0);
      float4 xa1 = *(const float4*)(xrow + kb0 + 4);
      float4 xb0 = *(const float4*)(xrow + kb1);
      float4 xb1 = *(const float4*)(xrow + kb1 + 4);
      v16bf a;
      a[0]  = (__bf16)(ma0.x * rcp + xa0.x);
      a[1]  = (__bf16)(ma0.y * rcp + xa0.y);
      a[2]  = (__bf16)(ma0.z * rcp + xa0.z);
      a[3]  = (__bf16)(ma0.w * rcp + xa0.w);
      a[4]  = (__bf16)(ma1.x * rcp + xa1.x);
      a[5]  = (__bf16)(ma1.y * rcp + xa1.y);
      a[6]  = (__bf16)(ma1.z * rcp + xa1.z);
      a[7]  = (__bf16)(ma1.w * rcp + xa1.w);
      a[8]  = (__bf16)(mb0.x * rcp + xb0.x);
      a[9]  = (__bf16)(mb0.y * rcp + xb0.y);
      a[10] = (__bf16)(mb0.z * rcp + xb0.z);
      a[11] = (__bf16)(mb0.w * rcp + xb0.w);
      a[12] = (__bf16)(mb1.x * rcp + xb1.x);
      a[13] = (__bf16)(mb1.y * rcp + xb1.y);
      a[14] = (__bf16)(mb1.z * rcp + xb1.z);
      a[15] = (__bf16)(mb1.w * rcp + xb1.w);
      #pragma unroll
      for (int ct = 0; ct < 8; ++ct) {
        v16bf b = *(const v16bf*)(&Bf[cv][kc4][ct][lane][0]);
        acc[ct] = __builtin_amdgcn_wmma_f32_16x16x32_bf16(
            false, a, false, b, (short)0, acc[ct], false, false);
      }
    }
  }

  // bias (per output column)
  #pragma unroll
  for (int ct = 0; ct < 8; ++ct) {
    float bb = b0[ct * 16 + lr];
    if (NCONV == 2) bb += b1[ct * 16 + lr];
    #pragma unroll
    for (int r = 0; r < 8; ++r) acc[ct][r] += bb;
  }

  float lw[8], lb[8];
  #pragma unroll
  for (int ct = 0; ct < 8; ++ct) { lw[ct] = lnw[ct * 16 + lr]; lb[ct] = lnb[ct * 16 + lr]; }

  // LayerNorm + ReLU + store.  D layout: VGPR r -> rows r (lanes 0-15) and r+8 (16-31).
  #pragma unroll
  for (int r = 0; r < 8; ++r) {
    float s1 = 0.0f, s2 = 0.0f;
    #pragma unroll
    for (int ct = 0; ct < 8; ++ct) { float v = acc[ct][r]; s1 += v; s2 += v * v; }
    #pragma unroll
    for (int m = 1; m < 16; m <<= 1) {  // stays within 16-lane half
      s1 += __shfl_xor(s1, m, 32);
      s2 += __shfl_xor(s2, m, 32);
    }
    float mu  = s1 * (1.0f / 128.0f);
    float var = s2 * (1.0f / 128.0f) - mu * mu;
    float rs  = rsqrtf(var + EPS_F);
    int orow  = m0 + r + half * 8;
    float* op = out + (size_t)orow * C_N;
    #pragma unroll
    for (int ct = 0; ct < 8; ++ct) {
      float v = (acc[ct][r] - mu) * rs * lw[ct] + lb[ct];
      op[ct * 16 + lr] = fmaxf(v, 0.0f);
    }
  }
}

// ------------------------------- launcher -----------------------------------
extern "C" void kernel_launch(void* const* d_in, const int* in_sizes, int n_in,
                              void* d_out, int out_size, void* d_ws, size_t ws_size,
                              hipStream_t stream) {
  const float* x_a      = (const float*)d_in[0];
  const float* x_b      = (const float*)d_in[1];
  const float* schema_x = (const float*)d_in[2];
  const int*   edge_ab  = (const int*)d_in[3];
  const int*   edge_ba  = (const int*)d_in[4];
  const int*   edge_aa  = (const int*)d_in[5];
  const int*   sei      = (const int*)d_in[6];
  const float* pre_W    = (const float*)d_in[7];
  const float* pre_b    = (const float*)d_in[8];
  const float* gcn_W    = (const float*)d_in[9];
  const float* gcn_b    = (const float*)d_in[10];
  const float* coeff_W  = (const float*)d_in[11];
  const float* coeff_b  = (const float*)d_in[12];
  const float* bases    = (const float*)d_in[13];
  const float* sage_b   = (const float*)d_in[14];
  const float* ln_w     = (const float*)d_in[15];
  const float* ln_b     = (const float*)d_in[16];

  float* out     = (float*)d_out;
  float* out_xa  = out;
  float* out_xb  = out + (size_t)NA_N * C_N;
  float* out_sf  = out + 2 * (size_t)NA_N * C_N;
  float* out_ori = out_sf + S_N * H_N;

  // workspace layout (floats)
  float* ws    = (float*)d_ws;
  float* sf_ws = ws;                               // 512
  float* Wmat  = sf_ws + 512;                      // 3 * 16384
  float* msg   = Wmat + 3 * 16384;                 // 3 * NA*C
  float* cnt   = msg + 3 * (size_t)NA_N * C_N;     // 3 * NA
  float* xa1   = cnt + 3 * NA_N;                   // NA*C
  float* xb1   = xa1 + (size_t)NA_N * C_N;         // NB*C

  const size_t NC = (size_t)NA_N * C_N;            // 6,400,000

  schema_kernel<<<1, 256, 0, stream>>>(schema_x, sei, pre_W, pre_b, gcn_W, gcn_b,
                                       sf_ws, out_sf, out_ori);

  const int scatter_grid = (E_N * 32) / 256;       // 100000
  const int gemm_grid    = (NA_N / 16 + 7) / 8;    // 391

  for (int l = 0; l < 2; ++l) {
    const float* xa_in  = (l == 0) ? x_a : xa1;
    const float* xb_in  = (l == 0) ? x_b : xb1;
    float*       xa_out = (l == 0) ? xa1 : out_xa;
    float*       xb_out = (l == 0) ? xb1 : out_xb;

    weight_kernel<<<3, 256, 0, stream>>>(sf_ws, coeff_W, coeff_b,
                                         bases + (size_t)l * NBASES * C_N * C_N, Wmat);

    zero_kernel<<<4096, 256, 0, stream>>>(msg, (long)(3 * NC + 3 * NA_N));

    scatter_kernel<<<scatter_grid, 256, 0, stream>>>(xa_in, edge_ab, msg + 0 * NC, cnt + 0 * NA_N); // dst b
    scatter_kernel<<<scatter_grid, 256, 0, stream>>>(xb_in, edge_ba, msg + 1 * NC, cnt + 1 * NA_N); // dst a
    scatter_kernel<<<scatter_grid, 256, 0, stream>>>(xa_in, edge_aa, msg + 2 * NC, cnt + 2 * NA_N); // dst a

    // out_b = LN(relu-wrapped) of sage(x_a -> x_b) with W_ab
    sage_gemm_ln_kernel<1><<<gemm_grid, 256, 0, stream>>>(
        msg + 0 * NC, cnt + 0 * NA_N, Wmat + 0 * 16384, sage_b + (l * 3 + 0) * C_N,
        nullptr, nullptr, nullptr, nullptr,
        xb_in, ln_w + (l * 2 + 1) * C_N, ln_b + (l * 2 + 1) * C_N, xb_out, NB_N);

    // out_a = sage(x_b -> x_a, W_ba) + sage(x_a -> x_a, W_aa)
    sage_gemm_ln_kernel<2><<<gemm_grid, 256, 0, stream>>>(
        msg + 1 * NC, cnt + 1 * NA_N, Wmat + 1 * 16384, sage_b + (l * 3 + 1) * C_N,
        msg + 2 * NC, cnt + 2 * NA_N, Wmat + 2 * 16384, sage_b + (l * 3 + 2) * C_N,
        xa_in, ln_w + (l * 2 + 0) * C_N, ln_b + (l * 2 + 0) * C_N, xa_out, NA_N);
  }
}
